// Processor_28518582846167
// MI455X (gfx1250) — matrix operand order repaired and checked by reading, not compiled, not run on hardware
//
#include <hip/hip_runtime.h>
#include <hip/hip_bf16.h>

typedef __attribute__((ext_vector_type(16))) _Float16 v16h;
typedef __attribute__((ext_vector_type(8)))  _Float16 v8h;
typedef __attribute__((ext_vector_type(8)))  float    v8f;

#define D_LAT 128
#define N_NODES 50000
#define N_EDGES 500000
#define LN_EPS 1e-5f

// ---------------- WMMA helpers ----------------

__device__ __forceinline__ v8f wmmaf(v16h a, v16h b, v8f c) {
    // D = A(16x32 f16) * B(32x16 f16) + C(16x16 f32)
    return __builtin_amdgcn_wmma_f32_16x16x32_f16(
        /*neg_a=*/false, a, /*neg_b=*/false, b,
        /*c_mod=*/(short)0, c, /*reuse_a=*/false, /*reuse_b=*/false);
}

// A-matrix fragment from LDS, row-major f16 rows (stride sa halfs).
// ISA 16-bit A 16x32 layout: lanes 0-15 hold M=lane, K={k0..k0+7, k0+16..k0+23};
// lanes 16-31 hold M=lane-16, K={k0+8..k0+15, k0+24..k0+31}.
__device__ __forceinline__ v16h load_a(const _Float16* A, int sa, int lane, int k0) {
    const int m  = lane & 15;
    const int hi = (lane >> 4) & 1;
    const _Float16* p = A + m * sa + k0 + hi * 8;
    union { v16h v; v8h p8[2]; } u;
    u.p8[0] = *(const v8h*)(p);
    u.p8[1] = *(const v8h*)(p + 16);
    return u.v;
}

// B-matrix fragment: B[k][n] = W[n][k], W f16 row-major with stride kpad.
// Lanes 0-15: column n = nbase+lane, K = {k0..k0+7, k0+16..k0+23} ; lanes 16-31: same column, K = {k0+8..k0+15, k0+24..k0+31}.
// => two 16-byte loads per lane.
__device__ __forceinline__ v16h load_b(const _Float16* W, int kpad, int nbase, int lane, int k0) {
    const int c  = lane & 15;
    const int hi = (lane >> 4) & 1;
    const _Float16* p = W + (size_t)(nbase + c) * kpad + k0 + hi * 8;
    union { v16h v; v8h p8[2]; } u;
    u.p8[0] = *(const v8h*)(p);
    u.p8[1] = *(const v8h*)(p + 16);
    return u.v;
}

// C/D f32 16x16 layout: VGPR r, lanes 0-15 -> M=r, N=lane; lanes 16-31 -> M=8+r, N=lane-16.
__device__ __forceinline__ void store_h(float* H, v8f acc, int nbase, int lane,
                                        const float* bias, bool relu) {
    const int c  = nbase + (lane & 15);
    const int hi = lane >> 4;
    const float bb = bias[c];
#pragma unroll
    for (int r = 0; r < 8; ++r) {
        float v = acc[r] + bb;
        if (relu) v = fmaxf(v, 0.0f);
        H[(r + 8 * hi) * D_LAT + c] = v;
    }
}

// One 16xK @ Kx128 GEMM slice: wave wv computes N-tiles {2wv, 2wv+1}.
__device__ __forceinline__ void gemm16(const _Float16* A, int sa, const _Float16* W,
                                       int kpad, int nch, float* H, const float* bias,
                                       bool relu, int lane, int wv) {
    v8f a0 = {};
    v8f a1 = {};
    const int n0 = wv * 32;
    const int n1 = wv * 32 + 16;
    for (int kc = 0; kc < nch; ++kc) {
        const int k0 = kc * 32;
        v16h av = load_a(A, sa, lane, k0);
        v16h b0 = load_b(W, kpad, n0, lane, k0);
        v16h b1 = load_b(W, kpad, n1, lane, k0);
        a0 = wmmaf(av, b0, a0);
        a1 = wmmaf(av, b1, a1);
    }
    store_h(H, a0, n0, lane, bias, relu);
    store_h(H, a1, n1, lane, bias, relu);
}

// LayerNorm stats over the 128 features of each of the 16 rows in H.
// 128 threads: thread t covers row t>>3, chunk (t&7)*16..+15.
__device__ __forceinline__ void ln_stats(const float* H, float* ps, float* ps2,
                                         float* mu, float* rs, int t) {
    const int r = t >> 3, g = t & 7;
    float s = 0.0f, s2 = 0.0f;
    const float* row = H + r * D_LAT + g * 16;
#pragma unroll
    for (int j = 0; j < 16; ++j) { float v = row[j]; s += v; s2 += v * v; }
    ps[t] = s; ps2[t] = s2;
    __syncthreads();
    if (t < 16) {
        float S = 0.0f, S2 = 0.0f;
#pragma unroll
        for (int g2 = 0; g2 < 8; ++g2) { S += ps[t * 8 + g2]; S2 += ps2[t * 8 + g2]; }
        const float m  = S * (1.0f / 128.0f);
        const float var = S2 * (1.0f / 128.0f) - m * m;
        mu[t] = m;
        rs[t] = rsqrtf(var + LN_EPS);
    }
    __syncthreads();
}

// ---------------- param plumbing ----------------

struct BlockP {
    const _Float16* W;   // f16, zero-padded to kpad
    const float* b;
    const float* beta;
    const float* g;
};
struct EdgeW { BlockP e1, e2, e3, n1; };
struct NodeW { BlockP n2, n3; };

// ---------------- kernels ----------------

__global__ void convw_kernel(const float* __restrict__ src, _Float16* __restrict__ dst,
                             int din, int kpad, int n) {
    int i = blockIdx.x * blockDim.x + threadIdx.x;
    if (i >= n) return;
    int o = i / kpad, k = i - o * kpad;
    dst[i] = (k < din) ? (_Float16)src[o * din + k] : (_Float16)0.0f;
}

__global__ void set_scalar_kernel(float* p, float v) { p[0] = v; }

// Fused edge pipeline: e1->e2->e3 (+residual e_new) then n1 + scatter into ssum/cnt.
__global__ __launch_bounds__(128) void edge_kernel(
        const float* __restrict__ X, const float* __restrict__ E,
        const int* __restrict__ EI, EdgeW w,
        float* __restrict__ Eout, float* __restrict__ ssum, float* __restrict__ cnt) {
    __shared__ __align__(16) _Float16 Abuf[16 * 424];   // [xr | xc | e | 1 | pad] K=416
    __shared__ __align__(16) _Float16 A2[16 * 128];
    __shared__ __align__(16) float H[16 * 128];
    __shared__ float ps[128], ps2[128];
    __shared__ float mu[16], rs[16];
    __shared__ int ridx[16], cidx[16];

    const int t = threadIdx.x;
    const int lane = t & 31;
    const int wv = t >> 5;
    const int ebase = blockIdx.x * 16;
    const int r8 = t >> 3, g8 = t & 7;

    if (t < 16) {
        ridx[t] = EI[ebase + t];
        cidx[t] = EI[N_EDGES + ebase + t];
    }
    __syncthreads();

    // Build A1 = [x[row] | x[col] | e | 1 | 0...] as f16
    for (int idx = t; idx < 16 * D_LAT; idx += 128) {
        const int r = idx >> 7, c = idx & 127;
        Abuf[r * 424 + c]             = (_Float16)X[(size_t)ridx[r] * D_LAT + c];
        Abuf[r * 424 + D_LAT + c]     = (_Float16)X[(size_t)cidx[r] * D_LAT + c];
        Abuf[r * 424 + 2 * D_LAT + c] = (_Float16)E[(size_t)(ebase + r) * D_LAT + c];
    }
    for (int idx = t; idx < 16 * 32; idx += 128) {
        const int r = idx >> 5, c = idx & 31;
        Abuf[r * 424 + 384 + c] = (c == 0) ? (_Float16)1.0f : (_Float16)0.0f;
    }
    __syncthreads();

    // ---- e1: K=416, ReLU + LN -> A2
    gemm16(Abuf, 424, w.e1.W, 416, 13, H, w.e1.b, true, lane, wv);
    __syncthreads();
    ln_stats(H, ps, ps2, mu, rs, t);
    {
        const float m = mu[r8], rr = rs[r8];
#pragma unroll
        for (int j = 0; j < 16; ++j) {
            const int c = g8 * 16 + j;
            A2[r8 * D_LAT + c] =
                (_Float16)((H[r8 * D_LAT + c] - m) * rr * w.e1.g[c] + w.e1.beta[c]);
        }
    }
    __syncthreads();

    // ---- e2: K=128, ReLU + LN -> A2
    gemm16(A2, 128, w.e2.W, 128, 4, H, w.e2.b, true, lane, wv);
    __syncthreads();
    ln_stats(H, ps, ps2, mu, rs, t);
    {
        const float m = mu[r8], rr = rs[r8];
#pragma unroll
        for (int j = 0; j < 16; ++j) {
            const int c = g8 * 16 + j;
            A2[r8 * D_LAT + c] =
                (_Float16)((H[r8 * D_LAT + c] - m) * rr * w.e2.g[c] + w.e2.beta[c]);
        }
    }
    __syncthreads();

    // ---- e3: K=128, LN (no relu), residual e_new = y + e ; store + stage f16 into Abuf[.,128..255]
    gemm16(A2, 128, w.e3.W, 128, 4, H, w.e3.b, false, lane, wv);
    __syncthreads();
    ln_stats(H, ps, ps2, mu, rs, t);
    {
        const float m = mu[r8], rr = rs[r8];
#pragma unroll
        for (int j = 0; j < 16; ++j) {
            const int c = g8 * 16 + j;
            const float y = (H[r8 * D_LAT + c] - m) * rr * w.e3.g[c] + w.e3.beta[c];
            const float enew = y + E[(size_t)(ebase + r8) * D_LAT + c];
            Eout[(size_t)(ebase + r8) * D_LAT + c] = enew;
            Abuf[r8 * 424 + D_LAT + c] = (_Float16)enew;   // n1 input = [x[row] | e_new]
        }
    }
    __syncthreads();

    // ---- n1: K=256, ReLU + LN, scatter-add into ssum / cnt
    gemm16(Abuf, 424, w.n1.W, 256, 8, H, w.n1.b, true, lane, wv);
    __syncthreads();
    ln_stats(H, ps, ps2, mu, rs, t);
    {
        const float m = mu[r8], rr = rs[r8];
        const size_t nb = (size_t)cidx[r8] * D_LAT;
#pragma unroll
        for (int j = 0; j < 16; ++j) {
            const int c = g8 * 16 + j;
            const float v = (H[r8 * D_LAT + c] - m) * rr * w.n1.g[c] + w.n1.beta[c];
            atomicAdd(&ssum[nb + c], v);
        }
    }
    if (t < 16) atomicAdd(&cnt[cidx[t]], 1.0f);
}

// Node pipeline: [x | agg | 1] -> n2 -> n3 -> +x
__global__ __launch_bounds__(128) void node_kernel(
        const float* __restrict__ X, const float* __restrict__ ssum,
        const float* __restrict__ cnt, NodeW w, float* __restrict__ Xout) {
    __shared__ __align__(16) _Float16 Abuf[16 * 296];   // K=288 padded
    __shared__ __align__(16) _Float16 A2[16 * 128];
    __shared__ __align__(16) float H[16 * 128];
    __shared__ float ps[128], ps2[128];
    __shared__ float mu[16], rs[16];

    const int t = threadIdx.x;
    const int lane = t & 31;
    const int wv = t >> 5;
    const int nbase = blockIdx.x * 16;
    const int r8 = t >> 3, g8 = t & 7;

    for (int idx = t; idx < 16 * D_LAT; idx += 128) {
        const int r = idx >> 7, c = idx & 127;
        const float ct = fmaxf(cnt[nbase + r], 1.0f);
        Abuf[r * 296 + c]         = (_Float16)X[(size_t)(nbase + r) * D_LAT + c];
        Abuf[r * 296 + D_LAT + c] = (_Float16)(ssum[(size_t)(nbase + r) * D_LAT + c] / ct);
    }
    for (int idx = t; idx < 16 * 32; idx += 128) {
        const int r = idx >> 5, c = idx & 31;
        Abuf[r * 296 + 256 + c] = (c == 0) ? (_Float16)1.0f : (_Float16)0.0f;
    }
    __syncthreads();

    // ---- n2: K=288, ReLU + LN -> A2
    gemm16(Abuf, 296, w.n2.W, 288, 9, H, w.n2.b, true, lane, wv);
    __syncthreads();
    ln_stats(H, ps, ps2, mu, rs, t);
    {
        const float m = mu[r8], rr = rs[r8];
#pragma unroll
        for (int j = 0; j < 16; ++j) {
            const int c = g8 * 16 + j;
            A2[r8 * D_LAT + c] =
                (_Float16)((H[r8 * D_LAT + c] - m) * rr * w.n2.g[c] + w.n2.beta[c]);
        }
    }
    __syncthreads();

    // ---- n3: K=128, LN, residual
    gemm16(A2, 128, w.n3.W, 128, 4, H, w.n3.b, false, lane, wv);
    __syncthreads();
    ln_stats(H, ps, ps2, mu, rs, t);
    {
        const float m = mu[r8], rr = rs[r8];
#pragma unroll
        for (int j = 0; j < 16; ++j) {
            const int c = g8 * 16 + j;
            const float y = (H[r8 * D_LAT + c] - m) * rr * w.n3.g[c] + w.n3.beta[c];
            Xout[(size_t)(nbase + r8) * D_LAT + c] = y + X[(size_t)(nbase + r8) * D_LAT + c];
        }
    }
}

// ---------------- host ----------------

static inline size_t alignup(size_t x) { return (x + 255) & ~(size_t)255; }

extern "C" void kernel_launch(void* const* d_in, const int* in_sizes, int n_in,
                              void* d_out, int out_size, void* d_ws, size_t ws_size,
                              hipStream_t stream) {
    (void)in_sizes; (void)n_in; (void)out_size; (void)ws_size;

    const float* X  = (const float*)d_in[0];
    const float* E  = (const float*)d_in[1];
    const int*   EI = (const int*)d_in[2];

    // Param order per jax pytree (sorted dict keys):
    // gn1,gn2 -> e1,e2,e3,n1,n2,n3 -> W,b,beta,g
    static const int DIN[6]  = {385, 128, 128, 256, 257, 128};
    static const int KPAD[6] = {416, 128, 128, 256, 288, 128};

    char* p = (char*)d_ws;
    _Float16* Wf[2][6];
    for (int L = 0; L < 2; ++L)
        for (int B = 0; B < 6; ++B) {
            Wf[L][B] = (_Float16*)p;
            p += alignup((size_t)D_LAT * KPAD[B] * sizeof(_Float16));
        }
    float* e1buf = (float*)p; p += alignup((size_t)N_EDGES * D_LAT * sizeof(float));
    float* x1buf = (float*)p; p += alignup((size_t)N_NODES * D_LAT * sizeof(float));
    float* ssum  = (float*)p; p += alignup((size_t)N_NODES * D_LAT * sizeof(float));
    float* cnt   = (float*)p; p += alignup((size_t)N_NODES * sizeof(float));

    // f32 -> padded f16 weight conversion (tiny; stays in L2)
    for (int L = 0; L < 2; ++L)
        for (int B = 0; B < 6; ++B) {
            const float* Wsrc = (const float*)d_in[3 + L * 24 + B * 4];
            const int n = D_LAT * KPAD[B];
            convw_kernel<<<(n + 255) / 256, 256, 0, stream>>>(Wsrc, Wf[L][B], DIN[B], KPAD[B], n);
        }

    auto mk = [&](int L, int B) -> BlockP {
        BlockP bp;
        bp.W    = Wf[L][B];
        bp.b    = (const float*)d_in[3 + L * 24 + B * 4 + 1];
        bp.beta = (const float*)d_in[3 + L * 24 + B * 4 + 2];
        bp.g    = (const float*)d_in[3 + L * 24 + B * 4 + 3];
        return bp;
    };

    float* out = (float*)d_out;
    const float* xin[2] = { X, x1buf };
    const float* ein[2] = { E, e1buf };
    float* eout[2] = { e1buf, out + (size_t)N_NODES * D_LAT };
    float* xout[2] = { x1buf, out };

    for (int L = 0; L < 2; ++L) {
        hipMemsetAsync(ssum, 0, (size_t)N_NODES * D_LAT * sizeof(float), stream);
        hipMemsetAsync(cnt, 0, (size_t)N_NODES * sizeof(float), stream);

        EdgeW ew; ew.e1 = mk(L, 0); ew.e2 = mk(L, 1); ew.e3 = mk(L, 2); ew.n1 = mk(L, 3);
        edge_kernel<<<N_EDGES / 16, 128, 0, stream>>>(xin[L], ein[L], EI, ew,
                                                      eout[L], ssum, cnt);

        NodeW nw; nw.n2 = mk(L, 4); nw.n3 = mk(L, 5);
        node_kernel<<<N_NODES / 16, 128, 0, stream>>>(xin[L], ssum, cnt, nw, xout[L]);
    }

    // u = ones((1,1))
    set_scalar_kernel<<<1, 1, 0, stream>>>(out + (size_t)N_NODES * D_LAT
                                               + (size_t)N_EDGES * D_LAT, 1.0f);
}
